// LDDMMShootingNetMap_32427003085492
// MI455X (gfx1250) — matrix operand order repaired and checked by reading, not compiled, not run on hardware
//
#include <hip/hip_runtime.h>

// ---------------- problem constants ----------------
#define NN 128
#define VOL (NN * NN * NN)        // 2097152
#define BATCH 2
#define CH 3
#define FIELD (BATCH * CH * VOL)  // 12582912 floats (~50.3 MB)
#define LINES (FIELD / NN)        // 98304
#define GROUPS (LINES / 16)       // 6144
#define LROW 132                  // padded LDS row (bank-conflict avoidance)
#define NB_SPATIAL ((BATCH * VOL) / 256)  // 16384
#define NB_FIELD (FIELD / 256)            // 49152
#define DTSTEP 0.1f
#define INVH 127.0f
#define INVH2 63.5f

typedef __attribute__((ext_vector_type(16))) __bf16 v16bf;
typedef __attribute__((ext_vector_type(8))) float v8f;
typedef union { v16bf v; uint4 q[2]; } bfu;
typedef int v4i __attribute__((vector_size(16)));  // matches builtin param type

#define AS1 __attribute__((address_space(1)))
#define AS3 __attribute__((address_space(3)))

#if __has_builtin(__builtin_amdgcn_global_load_async_to_lds_b32) && \
    __has_builtin(__builtin_amdgcn_global_load_async_to_lds_b128)
#define HAVE_ASYNC_LDS 1
#endif

#if __has_builtin(__builtin_amdgcn_s_wait_asynccnt)
#define WAIT_ASYNC() __builtin_amdgcn_s_wait_asynccnt(0)
#else
#define WAIT_ASYNC() asm volatile("s_wait_asynccnt 0" ::: "memory")
#endif

// Decompose a line index into its base element offset.
// l = bc*16384 + d2*128 + d3 ; base = bc*VOL + d2*mul2 + d3*mul3
__device__ __forceinline__ int lineBase(int l, int mul2, int mul3) {
  int bc = l >> 14;
  int r = l & 16383;
  return bc * VOL + (r >> 7) * mul2 + (r & 127) * mul3;
}

// ---------------------------------------------------------------------------
// Build the circular Gaussian operator G[i][j] = g[(i-j) mod 128], where
// g = IDFT( exp(-0.5*sigma^2*omega^2) )  (exactly the reference FFT filter).
// Stored as a bf16 hi/lo split for error-compensated WMMA.
// ---------------------------------------------------------------------------
__global__ void k_build_ops(unsigned short* __restrict__ Gh,
                            unsigned short* __restrict__ Gl) {
  int i = blockIdx.x;       // output row
  int j = threadIdx.x;      // input column
  int d = (i - j) & 127;
  const float sig = 0.05f;
  const float twopi = 6.283185307179586f;
  float acc = 0.0f;
  for (int k = 0; k < NN; ++k) {
    int kk = (k < 64) ? k : k - 128;
    float om = twopi * (float)kk * (127.0f / 128.0f);  // 2*pi*fftfreq(128, h)
    float filt = __expf(-0.5f * sig * sig * om * om);
    int ph = (k * d) & 127;  // range-reduced phase
    acc += filt * __cosf(twopi * (float)ph * (1.0f / 128.0f));
  }
  float g = acc * (1.0f / 128.0f);
  __bf16 hi = (__bf16)g;
  __bf16 lo = (__bf16)(g - (float)hi);
  Gh[i * NN + j] = __builtin_bit_cast(unsigned short, hi);
  Gl[i * NN + j] = __builtin_bit_cast(unsigned short, lo);
}

// ---------------------------------------------------------------------------
// Apply a 128x128 line operator along one axis of a [2,3,128,128,128] field
// via wave32 WMMA. One workgroup (8 waves) = 16 lines x all 128 positions.
// Error-compensated bf16: acc += Ghi*Bhi + Ghi*Blo + Glo*Bhi  (fp32 accum).
// ---------------------------------------------------------------------------
__global__ __launch_bounds__(256) void k_lineop(
    float* __restrict__ dst, const float* __restrict__ src,
    const unsigned short* __restrict__ Gh, const unsigned short* __restrict__ Gl,
    int estride, int mul2, int mul3) {
  __shared__ float Sbuf[16][LROW];  // 16 lines x 128 f32 (padded); reused for output
  const int t = threadIdx.x;
  const int l0 = blockIdx.x << 4;  // first line of this group

  // ---- stage 16 lines (f32) into LDS ----
#ifdef HAVE_ASYNC_LDS
  if (estride == 1) {
    // contiguous lines: 16B-granular async copies
    for (int qq = 0; qq < 2; ++qq) {
      int u = qq * 256 + t;        // 0..511
      int n = u & 15, k = (u >> 4) << 2;
      AS1 v4i* gp = (AS1 v4i*)(src + lineBase(l0 + n, mul2, mul3) + k);
      AS3 v4i* lp = (AS3 v4i*)&Sbuf[n][k];
      __builtin_amdgcn_global_load_async_to_lds_b128(gp, lp, 0, 0);
    }
  } else {
    // strided lines: per-element async copies (lane-addressed on both sides)
    for (int qq = 0; qq < 8; ++qq) {
      int u = qq * 256 + t;        // 0..2047
      int n = u & 15, k = u >> 4;
      AS1 int* gp = (AS1 int*)(src + lineBase(l0 + n, mul2, mul3) + k * estride);
      AS3 int* lp = (AS3 int*)&Sbuf[n][k];
      __builtin_amdgcn_global_load_async_to_lds_b32(gp, lp, 0, 0);
    }
  }
  WAIT_ASYNC();
#else
  for (int qq = 0; qq < 8; ++qq) {
    int u = qq * 256 + t;
    int n = u & 15, k = u >> 4;
    Sbuf[n][k] = src[lineBase(l0 + n, mul2, mul3) + k * estride];
  }
#endif
  __syncthreads();

  // ---- WMMA: wave w computes output rows [16w,16w+16) x 16 lines ----
  const int w = t >> 5;
  const int l = t & 31;
  const int half = l >> 4;   // K-half select (A/B fragment layouts)
  const int ln = l & 15;     // row (A) / line-column (B)
  const int row = (w << 4) + ln;

  v8f acc = {};
#pragma unroll
  for (int kc = 0; kc < 4; ++kc) {
    const int k0 = kc << 5;
    // A fragments straight from global (G is tiny and L0/L2-hot).
    // A layout: lane ln row=ln; elems 0..15 <- K=k0+half*16 .. +15
    bfu a_hi, a_lo;
    const unsigned short* grh = Gh + row * NN + k0 + half * 16;
    a_hi.q[0] = *(const uint4*)(grh);
    a_hi.q[1] = *(const uint4*)(grh + 8);
    const unsigned short* grl = Gl + row * NN + k0 + half * 16;
    a_lo.q[0] = *(const uint4*)(grl);
    a_lo.q[1] = *(const uint4*)(grl + 8);

    // B fragment: lane ln column=line ln; K = k0+half*16 .. +15 (contiguous in LDS)
    const float* rp = &Sbuf[ln][k0 + (half << 4)];
    float4 f0 = *(const float4*)(rp);
    float4 f1 = *(const float4*)(rp + 4);
    float4 f2 = *(const float4*)(rp + 8);
    float4 f3 = *(const float4*)(rp + 12);
    float vals[16];
    vals[0] = f0.x; vals[1] = f0.y; vals[2] = f0.z; vals[3] = f0.w;
    vals[4] = f1.x; vals[5] = f1.y; vals[6] = f1.z; vals[7] = f1.w;
    vals[8] = f2.x; vals[9] = f2.y; vals[10] = f2.z; vals[11] = f2.w;
    vals[12] = f3.x; vals[13] = f3.y; vals[14] = f3.z; vals[15] = f3.w;
    v16bf b_hi, b_lo;
#pragma unroll
    for (int e = 0; e < 16; ++e) {
      float f = vals[e];
      __bf16 h = (__bf16)f;
      b_hi[e] = h;
      b_lo[e] = (__bf16)(f - (float)h);
    }
    acc = __builtin_amdgcn_wmma_f32_16x16x32_bf16(false, a_hi.v, false, b_hi,
                                                  (short)0, acc, false, false);
    acc = __builtin_amdgcn_wmma_f32_16x16x32_bf16(false, a_hi.v, false, b_lo,
                                                  (short)0, acc, false, false);
    acc = __builtin_amdgcn_wmma_f32_16x16x32_bf16(false, a_lo.v, false, b_hi,
                                                  (short)0, acc, false, false);
  }
  __syncthreads();  // everyone done reading staged lines

  // D layout: VGPR r -> M = r + 8*half, N = ln. Scatter tile into LDS.
#pragma unroll
  for (int r = 0; r < 8; ++r)
    Sbuf[ln][(w << 4) + (half << 3) + r] = acc[r];
  __syncthreads();

  // ---- coalesced writeback ----
  if (estride == 1) {
    int n = t >> 4, i0 = (t & 15) << 3;
    float* dp = dst + lineBase(l0 + n, mul2, mul3) + i0;
    *(float4*)(dp) = *(const float4*)&Sbuf[n][i0];
    *(float4*)(dp + 4) = *(const float4*)&Sbuf[n][i0 + 4];
  } else {
    // mul3 == 1: adjacent lines are adjacent in memory
    int gb = lineBase(l0, mul2, mul3);
    int n = t & 15;
#pragma unroll
    for (int qq = 0; qq < 8; ++qq) {
      int i = (t >> 4) + (qq << 4);
      dst[gb + i * estride + n] = Sbuf[n][i];
    }
  }
}

// ---------------------------------------------------------------------------
// Pointwise / fused-stencil kernels (fp32 exact finite differences)
// ---------------------------------------------------------------------------
__device__ __forceinline__ float dsp(const float* f, int s, int q, int st) {
  if (q == 0) return (f[s + st] - f[s]) * INVH;
  if (q == 127) return (f[s] - f[s - st]) * INVH;
  return (f[s + st] - f[s - st]) * INVH2;
}

#define FOFF(b, c) ((((b) * 3 + (c)) << 21))

__global__ void k_zero(float* __restrict__ p) {
  p[blockIdx.x * 256 + threadIdx.x] = 0.0f;
}

__global__ void k_comb(float* __restrict__ d, const float* __restrict__ x,
                       const float* __restrict__ k, float s) {
  int i = blockIdx.x * 256 + threadIdx.x;
  d[i] = x[i] + s * k[i];
}

__global__ void k_copy(float* __restrict__ d, const float* __restrict__ s) {
  int i = blockIdx.x * 256 + threadIdx.x;
  d[i] = s[i];
}

// km[:,a] -= sum_j m[:,j] * d_a v_j
__global__ void k_rhs_term2(float* __restrict__ km, const float* __restrict__ m,
                            const float* __restrict__ v, int a, int shift, int st) {
  int t = blockIdx.x * 256 + threadIdx.x;
  int b = t >> 21, s = t & (VOL - 1);
  int q = (s >> shift) & 127;
  float acc = 0.0f;
#pragma unroll
  for (int j = 0; j < 3; ++j)
    acc += m[FOFF(b, j) + s] * dsp(v + FOFF(b, j), s, q, st);
  km[FOFF(b, a) + s] -= acc;
}

// kp[:,i] -= v[:,a] * d_a phi_i
__global__ void k_rhs_advect(float* __restrict__ kp, const float* __restrict__ phi,
                             const float* __restrict__ v, int a, int shift, int st) {
  int t = blockIdx.x * 256 + threadIdx.x;
  int b = t >> 21, s = t & (VOL - 1);
  int q = (s >> shift) & 127;
  float va = v[FOFF(b, a) + s];
#pragma unroll
  for (int i = 0; i < 3; ++i)
    kp[FOFF(b, i) + s] -= va * dsp(phi + FOFF(b, i), s, q, st);
}

// km[:,i] -= d_a ( m_i * v_a )   (conservative term; product formed on the fly)
__global__ void k_rhs_div(float* __restrict__ km, const float* __restrict__ m,
                          const float* __restrict__ v, int a, int shift, int st) {
  int t = blockIdx.x * 256 + threadIdx.x;
  int b = t >> 21, s = t & (VOL - 1);
  int q = (s >> shift) & 127;
  const float* vv = v + FOFF(b, a);
  float v0 = vv[s];
  float vp = (q < 127) ? vv[s + st] : 0.0f;
  float vm = (q > 0) ? vv[s - st] : 0.0f;
#pragma unroll
  for (int i = 0; i < 3; ++i) {
    const float* mm = m + FOFF(b, i);
    float d;
    if (q == 0)
      d = (mm[s + st] * vp - mm[s] * v0) * INVH;
    else if (q == 127)
      d = (mm[s] * v0 - mm[s - st] * vm) * INVH;
    else
      d = (mm[s + st] * vp - mm[s - st] * vm) * INVH2;
    km[FOFF(b, i) + s] -= d;
  }
}

// ---------------------------------------------------------------------------
// Host orchestration
// ---------------------------------------------------------------------------
extern "C" void kernel_launch(void* const* d_in, const int* in_sizes, int n_in,
                              void* d_out, int out_size, void* d_ws, size_t ws_size,
                              hipStream_t stream) {
  (void)in_sizes; (void)n_in; (void)out_size; (void)ws_size;
  const float* phi_in = (const float*)d_in[0];
  const float* m_in = (const float*)d_in[2];  // d_in[1] = I0_source (unused)
  float* phi = (float*)d_out;                 // phi state lives in d_out

  // workspace layout
  unsigned short* Gh = (unsigned short*)d_ws;
  unsigned short* Gl = Gh + NN * NN;
  float* fb = (float*)((char*)d_ws + 256 * 1024);
  float* m0 = fb + 0ull * FIELD;
  float* ms = fb + 1ull * FIELD;
  float* ps = fb + 2ull * FIELD;
  float* v  = fb + 3ull * FIELD;
  float* km = fb + 4ull * FIELD;
  float* kp = fb + 5ull * FIELD;
  float* am = fb + 6ull * FIELD;
  float* ap = fb + 7ull * FIELD;
  float* t1 = fb + 8ull * FIELD;
  float* t2 = fb + 9ull * FIELD;

  // per-axis line-op / stencil parameters (axis 0=X,1=Y,2=Z)
  const int est[3] = {16384, 128, 1};
  const int mu2[3] = {128, 16384, 16384};
  const int mu3[3] = {1, 1, 128};
  const int shf[3] = {14, 7, 0};

  k_build_ops<<<NN, NN, 0, stream>>>(Gh, Gl);
  k_copy<<<NB_FIELD, 256, 0, stream>>>(m0, m_in);
  k_copy<<<NB_FIELD, 256, 0, stream>>>(phi, phi_in);

  auto lineop = [&](float* dst, const float* src, int a) {
    k_lineop<<<GROUPS, 256, 0, stream>>>(dst, src, Gh, Gl, est[a], mu2[a], mu3[a]);
  };
  // km,kp <- RHS(msP, psP) with the EPDiff minus sign folded in
  auto stage = [&](const float* msP, const float* psP) {
    k_zero<<<NB_FIELD, 256, 0, stream>>>(km);
    k_zero<<<NB_FIELD, 256, 0, stream>>>(kp);
    lineop(t1, msP, 0);           // separable Gaussian: v = Gz Gy Gx m
    lineop(t2, t1, 1);
    lineop(v, t2, 2);
    for (int a = 0; a < 3; ++a) {
      k_rhs_term2<<<NB_SPATIAL, 256, 0, stream>>>(km, msP, v, a, shf[a], est[a]);
      k_rhs_advect<<<NB_SPATIAL, 256, 0, stream>>>(kp, psP, v, a, shf[a], est[a]);
      k_rhs_div<<<NB_SPATIAL, 256, 0, stream>>>(km, msP, v, a, shf[a], est[a]);
    }
  };
  auto comb = [&](float* d, const float* x, const float* k, float s) {
    k_comb<<<NB_FIELD, 256, 0, stream>>>(d, x, k, s);
  };

  const float hdt = 0.5f * DTSTEP;
  for (int step = 0; step < 10; ++step) {
    stage(m0, phi);                              // k1
    k_copy<<<NB_FIELD, 256, 0, stream>>>(am, km);
    k_copy<<<NB_FIELD, 256, 0, stream>>>(ap, kp);
    comb(ms, m0, km, hdt); comb(ps, phi, kp, hdt);
    stage(ms, ps);                               // k2
    comb(am, am, km, 2.0f); comb(ap, ap, kp, 2.0f);
    comb(ms, m0, km, hdt); comb(ps, phi, kp, hdt);
    stage(ms, ps);                               // k3
    comb(am, am, km, 2.0f); comb(ap, ap, kp, 2.0f);
    comb(ms, m0, km, DTSTEP); comb(ps, phi, kp, DTSTEP);
    stage(ms, ps);                               // k4
    comb(am, am, km, 1.0f); comb(ap, ap, kp, 1.0f);
    comb(m0, m0, am, DTSTEP / 6.0f);
    comb(phi, phi, ap, DTSTEP / 6.0f);           // phi state advances in d_out
  }
}
